// MultiHeadAreaAttention_45191645888913
// MI455X (gfx1250) — matrix operand order repaired and checked by reading, not compiled, not run on hardware
//
#include <hip/hip_runtime.h>
#include <stdint.h>

// ---------------------------------------------------------------------------
// MI455X (gfx1250) wave32 WMMA implementation of multi-head area attention.
// All matrix math on v_wmma_f32_16x16x32_bf16 (f32 accumulate).
// Pooled K (mean, pre-scaled) / V (sum) operands are precomputed in bf16 in
// WMMA fragment feed order and stay L2-resident (192MB L2 >> 24MB operands).
// GEMM tile staging uses global_load_async_to_lds (ASYNCcnt path).
// ---------------------------------------------------------------------------

typedef __attribute__((ext_vector_type(16))) __bf16 v16bf;
typedef __attribute__((ext_vector_type(8)))  float  v8f;
typedef int vint4 __attribute__((vector_size(16)));   // matches builtin V4i

#define AS1 __attribute__((address_space(1)))
#define AS3 __attribute__((address_space(3)))

#if defined(__has_builtin)
#if __has_builtin(__builtin_amdgcn_global_load_async_to_lds_b128)
#define USE_ASYNC_LDS 1
#endif
#endif

union Frag {                 // one WMMA 16-bit operand fragment per lane (32B)
    uint16_t s[16];
    uint4    q[2];
    v16bf    v;
};

__device__ __forceinline__ uint16_t f2bf(float f) {
    uint32_t u = __float_as_uint(f);
    u += 0x7FFFu + ((u >> 16) & 1u);
    return (uint16_t)(u >> 16);
}

__device__ __forceinline__ uint32_t f2bf2(float a, float b) {
#if defined(__has_builtin)
#if __has_builtin(__builtin_amdgcn_cvt_pk_bf16_f32)
    typedef __attribute__((ext_vector_type(2))) __bf16 bf16x2;
    union { bf16x2 v; uint32_t u; } x;
    x.v = __builtin_amdgcn_cvt_pk_bf16_f32(a, b);
    return x.u;
#else
    return (uint32_t)f2bf(a) | ((uint32_t)f2bf(b) << 16);
#endif
#else
    return (uint32_t)f2bf(a) | ((uint32_t)f2bf(b) << 16);
#endif
}

__device__ __forceinline__ void wait_async0() {
#if defined(USE_ASYNC_LDS)
#if __has_builtin(__builtin_amdgcn_s_wait_asynccnt)
    __builtin_amdgcn_s_wait_asynccnt(0);
#else
    asm volatile("s_wait_asynccnt 0x0" ::: "memory");
#endif
#endif
}

__device__ __forceinline__ v8f zero8() {
    v8f z = {0.f,0.f,0.f,0.f,0.f,0.f,0.f,0.f};
    return z;
}

// ---------------------------------------------------------------------------
// f32 -> bf16 conversion (2 elements per thread, packed cvt when available)
// ---------------------------------------------------------------------------
__global__ void cvt_bf16_kernel(const float* __restrict__ src,
                                uint16_t* __restrict__ dst, int n2) {
    int i = blockIdx.x * blockDim.x + threadIdx.x;
    if (i < n2) {
        float2 f = ((const float2*)src)[i];
        ((uint32_t*)dst)[i] = f2bf2(f.x, f.y);
    }
}

// ---------------------------------------------------------------------------
// C = A(MxK) @ B(NxK)^T + bias   (A,B bf16 row-major)
// out_mode: 0 = f32 row-major (MxN); 1 = f32 head layout (b,h,l,d);
//           2 = bf16 head layout (b,h,l,d)
// 256 threads = 8 waves; wave tile 32x32; WG tile 128(M) x 64(N); K step 32.
// ---------------------------------------------------------------------------
__global__ __launch_bounds__(256)
void gemm_bf16_kernel(const uint16_t* __restrict__ A,
                      const uint16_t* __restrict__ B,
                      const float* __restrict__ bias,
                      void* __restrict__ Cout,
                      int M, int N, int K, int out_mode) {
    __shared__ uint16_t ldsA[8][32 * 32];
    __shared__ uint16_t ldsB[8][32 * 32];

    const int tid  = threadIdx.x;
    const int wave = tid >> 5;
    const int lane = tid & 31;
    const int half = lane >> 4;
    const int l16  = lane & 15;
    const int wm   = wave >> 1;
    const int wn   = wave & 1;
    const int m0   = blockIdx.y * 128 + wm * 32;
    const int n0   = blockIdx.x * 64  + wn * 32;

    v8f acc[2][2];
    #pragma unroll
    for (int i = 0; i < 2; ++i)
        #pragma unroll
        for (int j = 0; j < 2; ++j) acc[i][j] = zero8();

    for (int kk = 0; kk < K; kk += 32) {
        __syncthreads();
        {   // stage 32x32 tiles (one row per lane, 64B each)
            const uint16_t* ga = A + (size_t)(m0 + lane) * K + kk;
            const uint16_t* gb = B + (size_t)(n0 + lane) * K + kk;
#if defined(USE_ASYNC_LDS)
            AS1 vint4* gA = (AS1 vint4*)(uintptr_t)ga;   // flat VA == global VA
            AS1 vint4* gB = (AS1 vint4*)(uintptr_t)gb;
            AS3 vint4* lA = (AS3 vint4*)&ldsA[wave][lane * 32];  // addrspacecast
            AS3 vint4* lB = (AS3 vint4*)&ldsB[wave][lane * 32];
            __builtin_amdgcn_global_load_async_to_lds_b128(gA, lA, 0,  0);
            __builtin_amdgcn_global_load_async_to_lds_b128(gA, lA, 16, 0);
            __builtin_amdgcn_global_load_async_to_lds_b128(gA, lA, 32, 0);
            __builtin_amdgcn_global_load_async_to_lds_b128(gA, lA, 48, 0);
            __builtin_amdgcn_global_load_async_to_lds_b128(gB, lB, 0,  0);
            __builtin_amdgcn_global_load_async_to_lds_b128(gB, lB, 16, 0);
            __builtin_amdgcn_global_load_async_to_lds_b128(gB, lB, 32, 0);
            __builtin_amdgcn_global_load_async_to_lds_b128(gB, lB, 48, 0);
            wait_async0();
#else
            const uint4* pa = (const uint4*)ga;
            uint4* da = (uint4*)&ldsA[wave][lane * 32];
            da[0] = pa[0]; da[1] = pa[1]; da[2] = pa[2]; da[3] = pa[3];
            const uint4* pb = (const uint4*)gb;
            uint4* db = (uint4*)&ldsB[wave][lane * 32];
            db[0] = pb[0]; db[1] = pb[1]; db[2] = pb[2]; db[3] = pb[3];
#endif
        }
        __syncthreads();

        Frag af[2], bf[2];
        #pragma unroll
        for (int mt = 0; mt < 2; ++mt) {
            // 16-bit A 16x32 layout: lanes<16 K=0..7,16..23 ; lanes>=16 K=8..15,24..31
            const uint16_t* base = &ldsA[wave][(mt * 16 + l16) * 32];
            af[mt].q[0] = *(const uint4*)(base + half * 8);
            af[mt].q[1] = *(const uint4*)(base + 16 + half * 8);
        }
        #pragma unroll
        for (int nt = 0; nt < 2; ++nt) {
            // B 32x16 layout: lane = column N, lanes<16 K=0..7,16..23 ; lanes>=16 K=8..15,24..31
            const uint16_t* base = &ldsB[wave][(nt * 16 + l16) * 32];
            bf[nt].q[0] = *(const uint4*)(base + half * 8);
            bf[nt].q[1] = *(const uint4*)(base + 16 + half * 8);
        }
        #pragma unroll
        for (int mt = 0; mt < 2; ++mt)
            #pragma unroll
            for (int nt = 0; nt < 2; ++nt)
                acc[mt][nt] = __builtin_amdgcn_wmma_f32_16x16x32_bf16(
                    false, af[mt].v, false, bf[nt].v,
                    (short)0, acc[mt][nt], false, false);
    }

    #pragma unroll
    for (int mt = 0; mt < 2; ++mt) {
        #pragma unroll
        for (int nt = 0; nt < 2; ++nt) {
            const int col = n0 + nt * 16 + l16;
            const float bv = bias ? bias[col] : 0.0f;
            #pragma unroll
            for (int r = 0; r < 8; ++r) {
                const int row = m0 + mt * 16 + half * 8 + r;
                const float val = acc[mt][nt][r] + bv;
                if (out_mode == 0) {
                    ((float*)Cout)[(size_t)row * N + col] = val;
                } else {
                    const int b = row >> 10, l = row & 1023;
                    const int h = col >> 6,  d = col & 63;
                    const size_t idx = ((size_t)(b * 16 + h) << 16) + (size_t)(l << 6) + d;
                    if (out_mode == 1) ((float*)Cout)[idx] = val;
                    else               ((uint16_t*)Cout)[idx] = f2bf(val);
                }
            }
        }
    }
}

// ---------------------------------------------------------------------------
// Area pooling: from f32 head-layout kh/vh produce bf16 WMMA-ready operands.
//   ka [w][bh][e][d] = mean_pool(k, w+1) * 1/8   (logit B fragments)
//   vaT[w][bh][d][e] = sum_pool(v, w+1)                   (ctx   B fragments)
// ---------------------------------------------------------------------------
__global__ __launch_bounds__(64)
void area_pool_kernel(const float* __restrict__ kh, const float* __restrict__ vh,
                      uint16_t* __restrict__ ka, uint16_t* __restrict__ vaT) {
    const int d  = threadIdx.x;                 // 0..63
    const int e  = blockIdx.x;                  // 0..1023
    const int bh = blockIdx.y;                  // 0..31
    const size_t hb = (size_t)bh << 16;         // bh * 1024 * 64
    const size_t W  = (size_t)32 << 16;         // per-width stride (2M elems)

    const size_t i0 = hb + (size_t)e * 64 + d;
    const float k0 = kh[i0];
    const float k1 = e >= 1 ? kh[i0 - 64]  : 0.f;
    const float k2 = e >= 2 ? kh[i0 - 128] : 0.f;
    const float v0 = vh[i0];
    const float v1 = e >= 1 ? vh[i0 - 64]  : 0.f;
    const float v2 = e >= 2 ? vh[i0 - 128] : 0.f;

    ka[0 * W + i0] = f2bf(k0 * 0.125f);
    ka[1 * W + i0] = f2bf((k0 + k1) * 0.0625f);
    ka[2 * W + i0] = f2bf((k0 + k1 + k2) * (0.125f / 3.0f));

    const size_t t0 = hb + (size_t)d * 1024 + e;
    vaT[0 * W + t0] = f2bf(v0);
    vaT[1 * W + t0] = f2bf(v0 + v1);
    vaT[2 * W + t0] = f2bf(v0 + v1 + v2);
}

// ---------------------------------------------------------------------------
// Fused area attention (flash-style, online softmax). One wave per
// (b,h, 16-query tile). All operand fragments are direct b128 loads of
// pre-pooled bf16 data; only softmax + P repack run on the VALU.
// ctx stored bf16 as (b, l, h*64+d) -- concat layout for the out-proj.
// ---------------------------------------------------------------------------
__global__ __launch_bounds__(32)
void area_attn_kernel(const uint16_t* __restrict__ qhb,
                      const uint16_t* __restrict__ ka,
                      const uint16_t* __restrict__ vaT,
                      uint16_t* __restrict__ ctxb,
                      const int* __restrict__ zero_pad_p) {
    __shared__ float pbuf[16 * 48];

    const int lane = threadIdx.x;
    const int half = lane >> 4;
    const int l16  = lane & 15;
    const int qt   = blockIdx.x;             // 0..63
    const int bh   = blockIdx.y;             // 0..31
    const int q0   = qt * 16;
    const size_t hb = (size_t)bh << 16;
    const size_t W  = (size_t)32 << 16;
    const int zp = *zero_pad_p;

    // ---- Q fragments (bf16 head layout, contiguous 16B chunks) ----
    Frag qa[2];
    {
        const uint16_t* qrow = qhb + hb + (size_t)(q0 + l16) * 64;
        #pragma unroll
        for (int c = 0; c < 2; ++c) {
            qa[c].q[0] = *(const uint4*)(qrow + c * 32 + half * 8);
            qa[c].q[1] = *(const uint4*)(qrow + c * 32 + 16 + half * 8);
        }
    }

    v8f O[4];
    #pragma unroll
    for (int nt = 0; nt < 4; ++nt) O[nt] = zero8();
    float mi[8], li[8];
    #pragma unroll
    for (int r = 0; r < 8; ++r) { mi[r] = -1e30f; li[r] = 0.0f; }

    for (int eb = 0; eb <= qt; ++eb) {
        const int e0 = eb * 16;

        // ---- logits per width: B fragments loaded straight from ka ----
        v8f lg[3];
        #pragma unroll
        for (int w = 0; w < 3; ++w) {
            const uint16_t* kbp = ka + (size_t)w * W + hb + (size_t)(e0 + l16) * 64;
            Frag kb0, kb1;
            kb0.q[0] = *(const uint4*)(kbp + half * 8);
            kb0.q[1] = *(const uint4*)(kbp + 16 + half * 8);
            kb1.q[0] = *(const uint4*)(kbp + 32 + half * 8);
            kb1.q[1] = *(const uint4*)(kbp + 32 + 16 + half * 8);
            v8f z = zero8();
            z = __builtin_amdgcn_wmma_f32_16x16x32_bf16(false, qa[0].v, false, kb0.v,
                                                        (short)0, z, false, false);
            z = __builtin_amdgcn_wmma_f32_16x16x32_bf16(false, qa[1].v, false, kb1.v,
                                                        (short)0, z, false, false);
            lg[w] = z;
        }

        // ---- mask + online softmax (16-lane xor reductions per C half) ----
        const int e = e0 + l16;
        #pragma unroll
        for (int r = 0; r < 8; ++r) {
            const int qrow = q0 + half * 8 + r;
            float rowmax = -1e30f;
            #pragma unroll
            for (int w = 0; w < 3; ++w) {
                float x = lg[w][r];
                const bool ok = (e <= qrow) && (e >= w);   // causal + area exists
                x = ok ? x : -1e30f;
                lg[w][r] = x;
                rowmax = fmaxf(rowmax, x);
            }
            #pragma unroll
            for (int off = 1; off < 16; off <<= 1)
                rowmax = fmaxf(rowmax, __shfl_xor(rowmax, off, 32));
            const float mnew  = fmaxf(mi[r], rowmax);
            const float alpha = __expf(mi[r] - mnew);
            mi[r] = mnew;
            float psum = 0.0f;
            #pragma unroll
            for (int w = 0; w < 3; ++w) {
                const float p = __expf(lg[w][r] - mnew);   // masked -> 0
                psum += p;
                pbuf[(half * 8 + r) * 48 + w * 16 + l16] = p;
            }
            #pragma unroll
            for (int off = 1; off < 16; off <<= 1)
                psum += __shfl_xor(psum, off, 32);
            li[r] = li[r] * alpha + psum;
            #pragma unroll
            for (int nt = 0; nt < 4; ++nt) O[nt][r] *= alpha;
        }
        __syncthreads();

        // ---- P transposed to A-fragments via LDS (48 areas, pad to 64) ----
        Frag pa0, pa1;
        {
            const float* pr = &pbuf[l16 * 48];
            #pragma unroll
            for (int j = 0; j < 8; ++j) pa0.s[j]     = f2bf(pr[half * 16 + j]);
            #pragma unroll
            for (int j = 0; j < 8; ++j) pa0.s[8 + j] = f2bf(pr[half * 16 + 8 + j]);
            #pragma unroll
            for (int j = 0; j < 8; ++j) pa1.s[j]     = half ? (uint16_t)0 : f2bf(pr[32 + j]);
            #pragma unroll
            for (int j = 0; j < 8; ++j) pa1.s[8 + j] = half ? (uint16_t)0 : f2bf(pr[40 + j]);   // half1: areas 48..63 pad
        }
        __syncthreads();

        // ---- ctx: O(16x64) += P(16x48) x Vsum(48x64), B frags from vaT ----
        #pragma unroll
        for (int nt = 0; nt < 4; ++nt) {
            const int d = nt * 16 + l16;
            Frag vb0, vb1;
            {   // chunk0: half0 -> w1 areas, half1 -> w2 areas (contiguous e)
                const uint16_t* p = vaT + (size_t)half * W + hb + (size_t)d * 1024 + e0;
                vb0.q[0] = ((const uint4*)p)[0];
                vb0.q[1] = ((const uint4*)p)[1];
            }
            if (half == 0) {   // chunk1: w3 areas; half1 rows are zero pad
                const uint16_t* p = vaT + 2 * W + hb + (size_t)d * 1024 + e0;
                vb1.q[0] = ((const uint4*)p)[0];
                vb1.q[1] = ((const uint4*)p)[1];
            } else {
                uint4 zq = {0u, 0u, 0u, 0u};
                vb1.q[0] = zq; vb1.q[1] = zq;
            }
            O[nt] = __builtin_amdgcn_wmma_f32_16x16x32_bf16(false, pa0.v, false, vb0.v,
                                                            (short)0, O[nt], false, false);
            O[nt] = __builtin_amdgcn_wmma_f32_16x16x32_bf16(false, pa1.v, false, vb1.v,
                                                            (short)0, O[nt], false, false);
        }
    }

    // ---- finalize: O / l, zero_pad row 0, store ctx bf16 (b, l, h*64+d) ----
    const int b = bh >> 4, h = bh & 15;
    #pragma unroll
    for (int r = 0; r < 8; ++r) {
        const int qrow = q0 + half * 8 + r;
        const float invl = li[r] > 0.0f ? 1.0f / li[r] : 0.0f;
        #pragma unroll
        for (int nt = 0; nt < 4; ++nt) {
            float val = O[nt][r] * invl;
            if (zp && qrow == 0) val = 0.0f;
            ctxb[(size_t)(b * 1024 + qrow) * 1024 + h * 64 + nt * 16 + l16] = f2bf(val);
        }
    }
}

// ---------------------------------------------------------------------------
// Host-side launch
// ---------------------------------------------------------------------------
extern "C" void kernel_launch(void* const* d_in, const int* in_sizes, int n_in,
                              void* d_out, int out_size, void* d_ws, size_t ws_size,
                              hipStream_t stream) {
    (void)in_sizes; (void)n_in; (void)out_size; (void)ws_size;

    const float* q  = (const float*)d_in[0];
    const float* k  = (const float*)d_in[1];
    const float* v  = (const float*)d_in[2];
    const float* Wq = (const float*)d_in[3];
    const float* bq = (const float*)d_in[4];
    const float* Wk = (const float*)d_in[5];
    const float* bk = (const float*)d_in[6];
    const float* Wv = (const float*)d_in[7];
    const float* bv = (const float*)d_in[8];
    const float* Wo = (const float*)d_in[9];
    const float* bo = (const float*)d_in[10];
    // d_in[11] = mask (causal tril; applied analytically in-kernel)
    const int* zero_pad = (const int*)d_in[12];

    char* ws = (char*)d_ws;
    const size_t MB = 1024 * 1024;
    uint16_t* qb   = (uint16_t*)(ws + 0 * MB);   // 4MB  bf16 q (GEMM A)
    uint16_t* kb   = (uint16_t*)(ws + 4 * MB);   // 4MB
    uint16_t* vb   = (uint16_t*)(ws + 8 * MB);   // 4MB
    uint16_t* Wqb  = (uint16_t*)(ws + 12 * MB);  // 2MB
    uint16_t* Wkb  = (uint16_t*)(ws + 14 * MB);  // 2MB
    uint16_t* Wvb  = (uint16_t*)(ws + 16 * MB);  // 2MB
    uint16_t* Wob  = (uint16_t*)(ws + 18 * MB);  // 2MB
    uint16_t* qhb  = (uint16_t*)(ws + 20 * MB);  // 4MB  bf16 (bh,l,d)
    float*    khf  = (float*)   (ws + 24 * MB);  // 8MB  f32  (bh,l,d)
    float*    vhf  = (float*)   (ws + 32 * MB);  // 8MB
    uint16_t* ka   = (uint16_t*)(ws + 40 * MB);  // 12MB bf16 pooled K (scaled means)
    uint16_t* vaT  = (uint16_t*)(ws + 52 * MB);  // 12MB bf16 pooled V (sums, d-major)
    uint16_t* ctxb = (uint16_t*)(ws + 64 * MB);  // 4MB  bf16 (b,l,h*64+d)

    const int M = 2048, N = 1024, K = 1024;
    const int nAct2 = M * K / 2;
    const int nW2   = N * K / 2;

    // 1) f32 -> bf16 (packed)
    cvt_bf16_kernel<<<(nAct2 + 255) / 256, 256, 0, stream>>>(q,  qb,  nAct2);
    cvt_bf16_kernel<<<(nAct2 + 255) / 256, 256, 0, stream>>>(k,  kb,  nAct2);
    cvt_bf16_kernel<<<(nAct2 + 255) / 256, 256, 0, stream>>>(v,  vb,  nAct2);
    cvt_bf16_kernel<<<(nW2   + 255) / 256, 256, 0, stream>>>(Wq, Wqb, nW2);
    cvt_bf16_kernel<<<(nW2   + 255) / 256, 256, 0, stream>>>(Wk, Wkb, nW2);
    cvt_bf16_kernel<<<(nW2   + 255) / 256, 256, 0, stream>>>(Wv, Wvb, nW2);
    cvt_bf16_kernel<<<(nW2   + 255) / 256, 256, 0, stream>>>(Wo, Wob, nW2);

    // 2) QKV projections (q -> bf16 head layout; k,v -> f32 head layout)
    dim3 gg(N / 64, M / 128), gb(256);
    gemm_bf16_kernel<<<gg, gb, 0, stream>>>(qb, Wqb, bq, qhb, M, N, K, 2);
    gemm_bf16_kernel<<<gg, gb, 0, stream>>>(kb, Wkb, bk, khf, M, N, K, 1);
    gemm_bf16_kernel<<<gg, gb, 0, stream>>>(vb, Wvb, bv, vhf, M, N, K, 1);

    // 3) area pooling -> bf16 WMMA-ready operands (L2 resident)
    area_pool_kernel<<<dim3(1024, 32), 64, 0, stream>>>(khf, vhf, ka, vaT);

    // 4) fused area attention -> ctx bf16 (b, l, h*64+d)
    area_attn_kernel<<<dim3(64, 32), 32, 0, stream>>>(qhb, ka, vaT, ctxb, zero_pad);

    // 5) output projection -> f32 d_out (b, l, D)
    gemm_bf16_kernel<<<gg, gb, 0, stream>>>(ctxb, Wob, bo, (float*)d_out, M, N, K, 0);
}
